// SSA_73203422593101
// MI455X (gfx1250) — matrix operand order repaired and checked by reading, not compiled, not run on hardware
//
#include <hip/hip_runtime.h>
#include <hip/hip_bf16.h>

// ---------------------------------------------------------------------------
// SSA forward for MI455X (gfx1250, wave32, WMMA 16x16x32 bf16)
// Shapes fixed by the reference: T=3 B=4 N=1024 C=512 H=8 D=64
// Uses (q k^T) v == q (k^T v)  (no softmax) to shrink attention to 64x64 KV.
// All matmuls run on v_wmma_f32_16x16x32_bf16; spike operands are exactly
// {0,1} so bf16 is exact for the KV / q@KV / projection GEMMs.
// ---------------------------------------------------------------------------

typedef __bf16 bf16_t;
typedef __attribute__((ext_vector_type(16))) __bf16 v16bf;
typedef __attribute__((ext_vector_type(8)))  float  v8f;

#define T_  3
#define B_  4
#define N_  1024
#define C_  512
#define H_  8
#define D_  64
#define TBN_ (T_ * B_ * N_)          // 12288 rows
#define SLICES_ (T_ * B_ * H_)       // 96 attention slices

__device__ __forceinline__ v8f wmma_bf16(v16bf a, v16bf b, v8f c) {
  return __builtin_amdgcn_wmma_f32_16x16x32_bf16(
      /*neg_a=*/false, a, /*neg_b=*/false, b,
      /*c_mod=*/(short)0, c, /*reuse_a=*/false, /*reuse_b=*/false);
}

// --- fragment loaders, contiguous-K order shared by A and B --------------
// A (16x32, MxK): lane L holds row (L&15); 16 contiguous K starting at
// (L>>4)*16.  B (32x16, KxN): lane L holds col (L&15); 16 contiguous K
// starting at (L>>4)*16.
__device__ __forceinline__ v16bf load_a_bf16(const bf16_t* X, int ldk, int row,
                                             int k0, int lane) {
  const bf16_t* p0 = X + (size_t)row * ldk + k0 + ((lane >> 4) * 16);
  const bf16_t* p1 = p0 + 8;
  v16bf r;
#pragma unroll
  for (int i = 0; i < 8; ++i) { r[i] = p0[i]; r[i + 8] = p1[i]; }
  return r;
}
__device__ __forceinline__ v16bf load_b_bf16(const bf16_t* Wm, int ldk, int col,
                                             int k0, int lane) {
  const bf16_t* p = Wm + (size_t)col * ldk + k0 + ((lane >> 4) * 16);
  v16bf r;
#pragma unroll
  for (int i = 0; i < 16; ++i) r[i] = p[i];
  return r;
}

// --- f32 -> bf16 conversion (weights once, activations once per branch) ----
__global__ __launch_bounds__(256) void cvt_bf16_kernel(
    const float* __restrict__ in, bf16_t* __restrict__ out, int n) {
  const int i = (blockIdx.x * blockDim.x + threadIdx.x) * 4;
  if (i + 3 < n) {
    const float4 v = *(const float4*)(in + i);
    out[i + 0] = (bf16_t)v.x;
    out[i + 1] = (bf16_t)v.y;
    out[i + 2] = (bf16_t)v.z;
    out[i + 3] = (bf16_t)v.w;
  }
}

// --- GEMM: Y[M,N] = X[M,K] @ W[N,K]^T + bias, f32 out ----------------------
// block = 256 thr (8 waves). Block tile 128x64, each wave one 16-row strip.
// W tile (64 cols x 32 K, bf16) double-buffered in LDS (2 x 4 KB): one
// barrier per K-step, stage of step k+1 overlaps the 4 WMMAs of step k.
// All 4 B fragments are pulled from LDS into registers before the WMMAs so a
// single dscnt wait covers them and the WMMAs issue back-to-back.
__global__ __launch_bounds__(256) void gemm_bias_kernel(
    const bf16_t* __restrict__ X, const bf16_t* __restrict__ Wb,
    const float* __restrict__ bias, float* __restrict__ Y,
    int M, int N, int K) {
  __shared__ bf16_t sB[2][64 * 32];  // [buf][col][k], 8 KB
  const int lane = threadIdx.x & 31;
  const int wave = threadIdx.x >> 5;
  const int mBase = blockIdx.x * 128 + wave * 16;
  const int nBase = blockIdx.y * 64;
  const int aRow = mBase + (lane & 15);
  const int trow = threadIdx.x >> 2;          // 0..63 (tile col)
  const int tchunk = (threadIdx.x & 3) * 8;   // 0,8,16,24 (k chunk)
  const bf16_t* wsrc = Wb + (size_t)(nBase + trow) * K + tchunk;
  v8f acc[4] = {};

  // prologue: stage K-step 0 into buffer 0
  *(ulonglong2*)&sB[0][trow * 32 + tchunk] = *(const ulonglong2*)wsrc;
  __syncthreads();

  const int nk = K / 32;
  for (int kt = 0; kt < nk; ++kt) {
    const int k0 = kt * 32;
    const int cur = kt & 1;
    // pull A (global) and all 4 B fragments (LDS) into registers
    v16bf a = load_a_bf16(X, K, aRow, k0, lane);
    v16bf bf[4];
#pragma unroll
    for (int j = 0; j < 4; ++j)
      bf[j] = *(const v16bf*)
          &sB[cur][(j * 16 + (lane & 15)) * 32 + (lane >> 4) * 16];
    // stage next K-step into the other buffer while WMMAs run
    if (kt + 1 < nk) {
      *(ulonglong2*)&sB[cur ^ 1][trow * 32 + tchunk] =
          *(const ulonglong2*)(wsrc + k0 + 32);
      __builtin_prefetch(X + (size_t)aRow * K + k0 + 32, 0, 1);
    }
#pragma unroll
    for (int j = 0; j < 4; ++j) acc[j] = wmma_bf16(a, bf[j], acc[j]);
    __syncthreads();
  }

  const int m0 = mBase + 8 * (lane >> 4);  // C/D layout: M = vgpr + 8*(lane>>4)
  const int nl = lane & 15;
#pragma unroll
  for (int j = 0; j < 4; ++j) {
    const int col = nBase + j * 16 + nl;
    const float bv = bias[col];
#pragma unroll
    for (int r = 0; r < 8; ++r)
      Y[(size_t)(m0 + r) * N + col] = acc[j][r] + bv;
  }
}

// --- BatchNorm statistics (two-pass, biased variance like torch BN) --------
__global__ __launch_bounds__(256) void bn_stats_kernel(
    const float* __restrict__ Y, float* __restrict__ sum,
    float* __restrict__ sumsq, int M, int N) {
  const int c = blockIdx.y * blockDim.x + threadIdx.x;  // channel
  float s = 0.f, sq = 0.f;
  for (int r = blockIdx.x; r < M; r += gridDim.x) {
    const float v = Y[(size_t)r * N + c];
    s += v; sq += v * v;
  }
  atomicAdd(&sum[c], s);
  atomicAdd(&sumsq[c], sq);
}

__global__ void bn_finalize_kernel(const float* __restrict__ sum,
                                   const float* __restrict__ sumsq,
                                   float* __restrict__ mean,
                                   float* __restrict__ rstd, float invM) {
  const int c = blockIdx.x * blockDim.x + threadIdx.x;
  const float m = sum[c] * invM;
  const float v = sumsq[c] * invM - m * m;
  mean[c] = m;
  rstd[c] = rsqrtf(v + 1e-5f);
}

// --- fused (optional BN) + LIF over time axis ------------------------------
// LAYOUT 0: bf16 spikes [t][b][h][n][d]   (q; WMMA-A friendly, contract d)
// LAYOUT 1: bf16 spikes [t][b][h][d][n]   (k,v; contract n contiguous)
// LAYOUT 2: bf16 spikes row-major [t*b*n][c]
// LAYOUT 3: f32  spikes row-major (final output)
template <int LAYOUT, bool HAS_BN>
__global__ __launch_bounds__(256) void bn_lif_kernel(
    const float* __restrict__ Y, const float* __restrict__ mean,
    const float* __restrict__ rstd, const float* __restrict__ gamma,
    const float* __restrict__ beta, void* __restrict__ out) {
  const int idx = blockIdx.x * blockDim.x + threadIdx.x;  // over B*N*C
  if (idx >= B_ * N_ * C_) return;
  const int c = idx % C_;
  const int bn = idx / C_;  // b*N + n
  const float g   = HAS_BN ? gamma[c] * rstd[c] : 1.f;
  const float off = HAS_BN ? beta[c] - mean[c] * g : 0.f;
  float mem = 0.f, spike = 0.f;
#pragma unroll
  for (int t = 0; t < T_; ++t) {
    const float x = Y[((size_t)t * B_ * N_ + bn) * C_ + c] * g + off;
    mem = mem * 0.25f * (1.f - spike) + x;   // first iter: mem = x0
    spike = (mem > 0.5f) ? 1.f : 0.f;
    if (LAYOUT <= 1) {
      const int h = c / D_, d = c % D_, n = bn % N_, b = bn / N_;
      const size_t sl = (size_t)(t * B_ + b) * H_ + h;
      const size_t o = (LAYOUT == 0) ? sl * (N_ * D_) + (size_t)n * D_ + d
                                     : sl * (D_ * N_) + (size_t)d * N_ + n;
      ((bf16_t*)out)[o] = (bf16_t)spike;
    } else {
      const size_t o = ((size_t)t * B_ * N_ + bn) * C_ + c;
      if (LAYOUT == 2) ((bf16_t*)out)[o] = (bf16_t)spike;
      else             ((float*)out)[o] = spike;
    }
  }
}

// --- KV = SCALE * k^T v per slice: [64,64], K-loop over N=1024 -------------
// Stores transposed (KVt[d2][d1], bf16) so the q@KV B-fragments are contiguous.
__global__ __launch_bounds__(256) void kv_kernel(
    const bf16_t* __restrict__ kt, const bf16_t* __restrict__ vt,
    bf16_t* __restrict__ kvt) {
  const int slice = blockIdx.x;  // (t*B+b)*H+h
  const bf16_t* Kp = kt + (size_t)slice * D_ * N_;
  const bf16_t* Vp = vt + (size_t)slice * D_ * N_;
  const int lane = threadIdx.x & 31, wave = threadIdx.x >> 5;
  const int mBase = (wave >> 1) * 16;       // d1 tile
  const int nBase = (wave & 1) * 32;        // d2 tile pair
  const int aRow = mBase + (lane & 15);
  v8f acc[2] = {};
  for (int k0 = 0; k0 < N_; k0 += 32) {
    v16bf a = load_a_bf16(Kp, N_, aRow, k0, lane);
#pragma unroll
    for (int j = 0; j < 2; ++j) {
      v16bf b = load_b_bf16(Vp, N_, nBase + j * 16 + (lane & 15), k0, lane);
      acc[j] = wmma_bf16(a, b, acc[j]);
    }
  }
  bf16_t* out = kvt + (size_t)slice * D_ * D_;
  const int m0 = mBase + 8 * (lane >> 4);
#pragma unroll
  for (int j = 0; j < 2; ++j) {
    const int col = nBase + j * 16 + (lane & 15);
#pragma unroll
    for (int r = 0; r < 8; ++r)
      out[(size_t)col * D_ + m0 + r] = (bf16_t)(acc[j][r] * 0.125f);
  }
}

// --- x = q @ KV per slice: [1024,64] = [1024,64]@[64,64] -------------------
// Writes f32 directly into [T,B,N,C] layout (c = h*64 + d2).
__global__ __launch_bounds__(256) void attn_out_kernel(
    const bf16_t* __restrict__ qs, const bf16_t* __restrict__ kvt,
    float* __restrict__ X) {
  const int slice = blockIdx.y;                 // 96
  const bf16_t* Qp = qs + (size_t)slice * N_ * D_;
  const bf16_t* KVp = kvt + (size_t)slice * D_ * D_;
  const int lane = threadIdx.x & 31, wave = threadIdx.x >> 5;
  const int mBase = blockIdx.x * 128 + wave * 16;  // n tile
  const int aRow = mBase + (lane & 15);
  v8f acc[4] = {};
  for (int k0 = 0; k0 < D_; k0 += 32) {
    v16bf a = load_a_bf16(Qp, D_, aRow, k0, lane);
#pragma unroll
    for (int j = 0; j < 4; ++j) {
      v16bf b = load_b_bf16(KVp, D_, j * 16 + (lane & 15), k0, lane);
      acc[j] = wmma_bf16(a, b, acc[j]);
    }
  }
  const int h = slice % H_;
  const int tb = slice / H_;  // t*B + b
  const int m0 = mBase + 8 * (lane >> 4);
#pragma unroll
  for (int j = 0; j < 4; ++j) {
    const int d2 = j * 16 + (lane & 15);
#pragma unroll
    for (int r = 0; r < 8; ++r)
      X[((size_t)tb * N_ + m0 + r) * C_ + h * D_ + d2] = acc[j][r];
  }
}

// ---------------------------------------------------------------------------
extern "C" void kernel_launch(void* const* d_in, const int* in_sizes, int n_in,
                              void* d_out, int out_size, void* d_ws,
                              size_t ws_size, hipStream_t stream) {
  (void)in_sizes; (void)n_in; (void)out_size; (void)ws_size;
  // Input order: q,k,v, [Wq,bq,gq,betaq], [Wk,..], [Wv,..], [Wp,bp,gp,betap]
  const float* qkv[3] = {(const float*)d_in[0], (const float*)d_in[1],
                         (const float*)d_in[2]};

  // Workspace layout (offsets in bytes, 256-aligned)
  char* ws = (char*)d_ws;
  float*  fbuf = (float*)(ws + 0);              // 25.2 MB f32 scratch (Y / x)
  bf16_t* qs   = (bf16_t*)(ws + 25165824);      // 12.6 MB
  bf16_t* ks   = (bf16_t*)(ws + 37748736);      // 12.6 MB
  bf16_t* vs   = (bf16_t*)(ws + 50331648);      // 12.6 MB
  bf16_t* s2   = (bf16_t*)(ws + 62914560);      // 12.6 MB (Xbf early, spikes2 late)
  bf16_t* kvt  = (bf16_t*)(ws + 75497472);      // 0.79 MB
  bf16_t* wbf  = (bf16_t*)(ws + 76283904);      // 0.5 MB bf16 weights
  float*  sum   = (float*)(ws + 76808192);
  float*  sumsq = (float*)(ws + 76810496);
  float*  meanp = (float*)(ws + 76812800);
  float*  rstdp = (float*)(ws + 76815104);

  const dim3 gGemm(TBN_ / 128, C_ / 64);   // 96 x 8
  const dim3 gStats(64, C_ / 256);         // 64 x 2
  const int  nElem = B_ * N_ * C_;         // 2M
  const int  gLif = (nElem + 255) / 256;
  const int  nX = TBN_ * C_;               // 6.29M activations
  const int  nW = C_ * C_;                 // 262k weights
  const float invM = 1.0f / (float)TBN_;

  // --- three branches: linear -> BN stats -> BN+LIF -> spike layout --------
  for (int br = 0; br < 3; ++br) {
    const float* W  = (const float*)d_in[3 + br * 4 + 0];
    const float* bb = (const float*)d_in[3 + br * 4 + 1];
    const float* gg = (const float*)d_in[3 + br * 4 + 2];
    const float* be = (const float*)d_in[3 + br * 4 + 3];
    cvt_bf16_kernel<<<nX / 1024, 256, 0, stream>>>(qkv[br], s2, nX);
    cvt_bf16_kernel<<<nW / 1024, 256, 0, stream>>>(W, wbf, nW);
    gemm_bias_kernel<<<gGemm, 256, 0, stream>>>(s2, wbf, bb, fbuf, TBN_, C_,
                                                C_);
    hipMemsetAsync(sum, 0, 2 * C_ * sizeof(float), stream);
    bn_stats_kernel<<<gStats, 256, 0, stream>>>(fbuf, sum, sumsq, TBN_, C_);
    bn_finalize_kernel<<<C_ / 256, 256, 0, stream>>>(sum, sumsq, meanp, rstdp,
                                                     invM);
    if (br == 0)
      bn_lif_kernel<0, true><<<gLif, 256, 0, stream>>>(fbuf, meanp, rstdp, gg,
                                                       be, qs);
    else if (br == 1)
      bn_lif_kernel<1, true><<<gLif, 256, 0, stream>>>(fbuf, meanp, rstdp, gg,
                                                       be, ks);
    else
      bn_lif_kernel<1, true><<<gLif, 256, 0, stream>>>(fbuf, meanp, rstdp, gg,
                                                       be, vs);
  }

  // --- attention via KV = SCALE * k^T v, then x = q @ KV -------------------
  kv_kernel<<<SLICES_, 256, 0, stream>>>(ks, vs, kvt);
  attn_out_kernel<<<dim3(N_ / 128, SLICES_), 256, 0, stream>>>(qs, kvt, fbuf);

  // --- LIF -> projection linear -> BN -> final LIF -> d_out ----------------
  bn_lif_kernel<2, false><<<gLif, 256, 0, stream>>>(fbuf, nullptr, nullptr,
                                                    nullptr, nullptr, s2);
  const float* Wp = (const float*)d_in[15];
  const float* bp = (const float*)d_in[16];
  const float* gp = (const float*)d_in[17];
  const float* betap = (const float*)d_in[18];
  cvt_bf16_kernel<<<nW / 1024, 256, 0, stream>>>(Wp, wbf, nW);
  gemm_bias_kernel<<<gGemm, 256, 0, stream>>>(s2, wbf, bp, fbuf, TBN_, C_, C_);
  hipMemsetAsync(sum, 0, 2 * C_ * sizeof(float), stream);
  bn_stats_kernel<<<gStats, 256, 0, stream>>>(fbuf, sum, sumsq, TBN_, C_);
  bn_finalize_kernel<<<C_ / 256, 256, 0, stream>>>(sum, sumsq, meanp, rstdp,
                                                   invM);
  bn_lif_kernel<3, true><<<gLif, 256, 0, stream>>>(fbuf, meanp, rstdp, gp,
                                                   betap, (float*)d_out);
}
